// GraphModel_89885075570937
// MI455X (gfx1250) — matrix operand order repaired and checked by reading, not compiled, not run on hardware
//
#include <hip/hip_runtime.h>

// ---------------------------------------------------------------------------
// Problem constants
// ---------------------------------------------------------------------------
#define KT   12
#define NN   65536
#define DIN  128
#define D2   256
#define D3   128
#define D4   256
#define NC   16

#define ROWS   64          // node rows per workgroup (4 waves x 16)
#define WGSZ   128         // 4 wave32

// packed-weight layout (per k, element offsets into bf16 pack)
#define PK_L1   0          // 128x256 -> 32768 elems
#define PK_L2   32768      // 256x128 -> 32768
#define PK_L3   65536      // 128x128 -> 16384
#define PK_L4   81920      // 128x128 -> 16384
#define PK_PERK 98304
#define PK_TOTAL (KT * PK_PERK)      // 1,179,648 bf16 elems
#define WS_PHI_BYTES 8192            // phi[1536] f32 lives at d_ws+0

typedef __attribute__((ext_vector_type(16))) __bf16 v16bf;
typedef __attribute__((ext_vector_type(8)))  __bf16 v8bf;
typedef __attribute__((ext_vector_type(8)))  float  v8f;

union bfpack16 { v16bf v; v8bf h[2]; };

__device__ __forceinline__ __bf16 f2bf(float f) {
    return (__bf16)f;                 // native RNE fptrunc -> hw bf16 cvt
}

__device__ __forceinline__ float lrelu(float v) {
    return fmaxf(v, 0.01f * v);       // == leaky_relu(v, 0.01) for all v
}

// ---------------------------------------------------------------------------
// Kernel 0: pack fp32 weights into WMMA bf16 B-fragment order, zero phi.
// B-frag layout (16-bit, 32x16): lane L -> col N = L&15, K = (L>>4)*8 + (e&7) + (e>>3)*16,
// e = 0..15 in two 8-runs, 16 apart in K.  Frag flat index:
//   ((ntile*ktc + ktile)*32 + lane)*16 + e
// ---------------------------------------------------------------------------
__global__ __launch_bounds__(256)
void prep_kernel(const float* __restrict__ W1, const float* __restrict__ W2,
                 const float* __restrict__ W3, const float* __restrict__ W4,
                 __bf16* __restrict__ wpk, float* __restrict__ phi)
{
    int idx = blockIdx.x * 256 + threadIdx.x;
    if (idx < KT * D3) phi[idx] = 0.0f;
    if (idx >= PK_TOTAL) return;

    int k = idx / PK_PERK;
    int r = idx - k * PK_PERK;

    const float* W; int Ncol, ktc, base;
    if (r < PK_L2)      { W = W1 + (size_t)k*DIN*D2; Ncol=D2; ktc=4; base=PK_L1; }
    else if (r < PK_L3) { W = W2 + (size_t)k*D2*D3;  Ncol=D3; ktc=8; base=PK_L2; }
    else if (r < PK_L4) { W = W3 + (size_t)k*D3*D3;  Ncol=D3; ktc=4; base=PK_L3; }
    else                { W = W4 + (size_t)k*D3*D3;  Ncol=D3; ktc=4; base=PK_L4; }

    int e_rel = r - base;
    int frag  = e_rel >> 9;          // /512
    int lane  = (e_rel >> 4) & 31;
    int el    = e_rel & 15;
    int ktile = frag % ktc;
    int ntile = frag / ktc;
    int Kidx  = ktile * 32 + (lane >> 4) * 8 + (el & 7) + (el >> 3) * 16;
    int Nidx  = ntile * 16 + (lane & 15);
    wpk[idx] = f2bf(W[(size_t)Kidx * Ncol + Nidx]);
}

// ---------------------------------------------------------------------------
// One fused MLP layer on a 16-row strip owned by this wave.
//  A-frag (16-bit 16x32): lane L -> row M = L&15; K = kt*32 + (L>>4)*8 + {0..7}
//  and + 16 + {0..7}  -> two 16B LDS loads.
//  C/D layout: VGPR r -> row = r + 8*(L>>4), col = L&15.
// First WMMA of each chain uses inline-0 C operand (acc = {}), bias is added
// in the epilogue (v_pk_add) -- avoids accumulator-init movs + hazard NOPs.
// ---------------------------------------------------------------------------
__device__ __forceinline__
void mlp_layer(const __bf16* __restrict__ in, int Kd,
               __bf16* __restrict__ out, int Ncol,
               const __bf16* __restrict__ wfrag, int ktc,
               const float* __restrict__ bias,
               int wave, int lane, bool last, float* __restrict__ phi_s)
{
    const int mrow = lane & 15;
    const int kgrp = lane >> 4;
    const int row  = wave * 16 + mrow;
    const int ntiles = Ncol >> 4;

    for (int nt = 0; nt < ntiles; ++nt) {
        v8f acc = {};                              // inline-0 C on first WMMA

        const __bf16* wq = wfrag + ((size_t)(nt * ktc) * 32 + lane) * 16;
        const __bf16* ap = in + (size_t)row * Kd + kgrp * 8;
        for (int kt = 0; kt < ktc; ++kt) {
            bfpack16 a;
            a.h[0] = *(const v8bf*)(ap);          // K = kt*32 + kgrp*8 + 0..7
            a.h[1] = *(const v8bf*)(ap + 16);     // K = kt*32 + kgrp*8 + 16..23
            v16bf b = *(const v16bf*)(wq);
            acc = __builtin_amdgcn_wmma_f32_16x16x32_bf16(
                      false, a.v, false, b, (short)0, acc, false, false);
            ap += 32;
            wq += 512;                            // next K-tile fragment
        }

        const int col = nt * 16 + mrow;
        const float bv = bias[col];
        if (!last) {
            const int rbase = wave * 16 + kgrp * 8;
#pragma unroll
            for (int r = 0; r < 8; ++r) {
                out[(size_t)(rbase + r) * Ncol + col] = f2bf(lrelu(acc[r] + bv));
            }
        } else {
            float p = 0.0f;
#pragma unroll
            for (int r = 0; r < 8; ++r) p += lrelu(acc[r] + bv);
            p += __shfl_xor(p, 16, 32);           // fold rows 0-7 with 8-15
            if (kgrp == 0) atomicAdd(&phi_s[col], p);
        }
    }
}

// ---------------------------------------------------------------------------
// Kernel 1: fused per-hop 4-layer MLP + node reduction.
// grid = (NN/ROWS, KT), block = 128 (4 waves); each wave owns 16 rows.
// ---------------------------------------------------------------------------
__global__ __launch_bounds__(WGSZ)
void hop_mlp_kernel(const float* __restrict__ x,
                    const __bf16* __restrict__ wpk,
                    const float* __restrict__ b1, const float* __restrict__ b2,
                    const float* __restrict__ b3, const float* __restrict__ b4,
                    float* __restrict__ phi)
{
    __shared__ __align__(16) __bf16 bufA[ROWS * 128];   // x / h2 (width 128)
    __shared__ __align__(16) __bf16 bufB[ROWS * 256];   // h1 (256) / h3 (128)
    __shared__ float phi_s[D3];

    const int k    = blockIdx.y;
    const int tile = blockIdx.x;
    const int tid  = threadIdx.x;
    const int wave = tid >> 5;
    const int lane = tid & 31;

    if (tid < D3) phi_s[tid] = 0.0f;

    // cooperative load: ROWS x DIN fp32 -> bf16 LDS, coalesced float4
    const float* xs = x + ((size_t)k * NN + (size_t)tile * ROWS) * DIN;
    for (int i = tid; i < ROWS * DIN / 4; i += WGSZ) {
        float4 v = ((const float4*)xs)[i];
        int o = i * 4;
        bufA[o + 0] = f2bf(v.x); bufA[o + 1] = f2bf(v.y);
        bufA[o + 2] = f2bf(v.z); bufA[o + 3] = f2bf(v.w);
    }
    __syncthreads();

    const __bf16* wk = wpk + (size_t)k * PK_PERK;
    mlp_layer(bufA, DIN, bufB, D2,  wk + PK_L1, 4, b1 + k * D2, wave, lane, false, nullptr);
    mlp_layer(bufB, D2,  bufA, D3,  wk + PK_L2, 8, b2 + k * D3, wave, lane, false, nullptr);
    mlp_layer(bufA, D3,  bufB, D3,  wk + PK_L3, 4, b3 + k * D3, wave, lane, false, nullptr);
    mlp_layer(bufB, D3,  nullptr, D3, wk + PK_L4, 4, b4 + k * D3, wave, lane, true, phi_s);

    __syncthreads();
    if (tid < D3) atomicAdd(&phi[k * D3 + tid], phi_s[tid]);
}

// ---------------------------------------------------------------------------
// Kernel 2: tiny decoder, one block, fp32 scalar.
// ---------------------------------------------------------------------------
__global__ __launch_bounds__(256)
void decoder_kernel(const float* __restrict__ phi,
                    const float* __restrict__ DW1, const float* __restrict__ Db1,
                    const float* __restrict__ DW2, const float* __restrict__ Db2,
                    const float* __restrict__ DW3, const float* __restrict__ Db3,
                    const float* __restrict__ DW4, const float* __restrict__ Db4,
                    float* __restrict__ out)
{
    __shared__ float phis[KT * D3];
    __shared__ float za[D4];
    __shared__ float zb[D4];
    const int d = threadIdx.x;

    for (int i = d; i < KT * D3; i += 256) phis[i] = phi[i];
    __syncthreads();

    float s = Db1[d];
    for (int i = 0; i < KT * D3; ++i) s += phis[i] * DW1[(size_t)i * D4 + d];
    za[d] = lrelu(s);
    __syncthreads();

    s = Db2[d];
    for (int i = 0; i < D4; ++i) s += za[i] * DW2[i * D4 + d];
    zb[d] = lrelu(s);
    __syncthreads();

    s = Db3[d];
    for (int i = 0; i < D4; ++i) s += zb[i] * DW3[i * D4 + d];
    za[d] = lrelu(s);
    __syncthreads();

    if (d < NC) {
        s = Db4[d];
        for (int i = 0; i < D4; ++i) s += za[i] * DW4[i * NC + d];
        out[d] = 1.0f / (1.0f + expf(-s));
    }
}

// ---------------------------------------------------------------------------
// Host launcher
// ---------------------------------------------------------------------------
extern "C" void kernel_launch(void* const* d_in, const int* in_sizes, int n_in,
                              void* d_out, int out_size, void* d_ws, size_t ws_size,
                              hipStream_t stream)
{
    const float* x   = (const float*)d_in[0];
    const float* W1  = (const float*)d_in[1];
    const float* b1  = (const float*)d_in[2];
    const float* W2  = (const float*)d_in[3];
    const float* b2  = (const float*)d_in[4];
    const float* W3  = (const float*)d_in[5];
    const float* b3  = (const float*)d_in[6];
    const float* W4  = (const float*)d_in[7];
    const float* b4  = (const float*)d_in[8];
    const float* DW1 = (const float*)d_in[9];
    const float* Db1 = (const float*)d_in[10];
    const float* DW2 = (const float*)d_in[11];
    const float* Db2 = (const float*)d_in[12];
    const float* DW3 = (const float*)d_in[13];
    const float* Db3 = (const float*)d_in[14];
    const float* DW4 = (const float*)d_in[15];
    const float* Db4 = (const float*)d_in[16];

    float*  phi = (float*)d_ws;
    __bf16* wpk = (__bf16*)((char*)d_ws + WS_PHI_BYTES);

    // 1) pack weights to bf16 fragments + zero phi
    int prep_blocks = (PK_TOTAL + 255) / 256;                 // 4608
    prep_kernel<<<prep_blocks, 256, 0, stream>>>(W1, W2, W3, W4, wpk, phi);

    // 2) fused hop MLP chain + node sums
    dim3 grid(NN / ROWS, KT);
    hop_mlp_kernel<<<grid, WGSZ, 0, stream>>>(x, wpk, b1, b2, b3, b4, phi);

    // 3) decoder
    decoder_kernel<<<1, 256, 0, stream>>>(phi, DW1, Db1, DW2, Db2,
                                          DW3, Db3, DW4, Db4, (float*)d_out);
}
